// Attention_19636590478064
// MI455X (gfx1250) — matrix operand order repaired and checked by reading, not compiled, not run on hardware
//
#include <hip/hip_runtime.h>

#define DIMV 4096
#define NH   32
#define HD   128
#define SEQV 2048
#define PREV 30
#define PREP 32
#define HALFV 64

typedef __bf16 bf16_t;
typedef __attribute__((ext_vector_type(16))) __bf16 v16bf;
typedef __attribute__((ext_vector_type(8)))  float  v8f;
typedef __attribute__((ext_vector_type(4)))  unsigned int u32x4;
typedef __attribute__((ext_vector_type(8)))  int i32x8;
typedef __attribute__((ext_vector_type(4)))  int i32x4;

union Frag {
    v16bf v;
    u32x4 q[2];
};

static __device__ __forceinline__ v8f v8f_zero() {
    v8f z = {0.f, 0.f, 0.f, 0.f, 0.f, 0.f, 0.f, 0.f};
    return z;
}

static __device__ __forceinline__ v8f wmma_bf16(const Frag& a, const Frag& b, v8f c) {
    return __builtin_amdgcn_wmma_f32_16x16x32_bf16(false, a.v, false, b.v,
                                                   (short)0, c, false, false);
}

static __device__ __forceinline__ float hmax16(float v) {
    v = fmaxf(v, __shfl_xor(v, 1, 32));
    v = fmaxf(v, __shfl_xor(v, 2, 32));
    v = fmaxf(v, __shfl_xor(v, 4, 32));
    v = fmaxf(v, __shfl_xor(v, 8, 32));
    return v;
}
static __device__ __forceinline__ float hsum16(float v) {
    v += __shfl_xor(v, 1, 32);
    v += __shfl_xor(v, 2, 32);
    v += __shfl_xor(v, 4, 32);
    v += __shfl_xor(v, 8, 32);
    return v;
}

// ---- CDNA5 async copy: global -> LDS, per-lane, tracked by ASYNCcnt ----
static __device__ __forceinline__ void async_b128(unsigned lds_off, const void* gp) {
    asm volatile("global_load_async_to_lds_b128 %0, %1, off"
                 :: "v"(lds_off), "v"((unsigned long long)(uintptr_t)gp)
                 : "memory");
}
static __device__ __forceinline__ void wait_async0() {
    asm volatile("s_wait_asynccnt 0x0" ::: "memory");
}

// ---- CDNA5 Tensor Data Mover: DMA a rows x cols bf16 tile into LDS ----
// 2D descriptor: data_size=2B, tile_dim0=cols, tile_dim1=rows, dim0 stride = stride_elems.
static __device__ __forceinline__ void tdm_load_tile(const bf16_t* gsrc, unsigned lds_off,
                                                     int rows, int cols, int stride_elems) {
    unsigned long long ga = (unsigned long long)(uintptr_t)gsrc;
    u32x4 g0;
    g0.x = 1u;                                           // count=1, user descriptor
    g0.y = lds_off;                                      // lds_addr
    g0.z = (unsigned)(ga & 0xffffffffu);                 // global_addr lo
    g0.w = (unsigned)((ga >> 32) & 0x01ffffffu) | (2u << 30);  // global_addr hi | type=2
    unsigned td0 = (unsigned)stride_elems;               // tensor_dim0 (elements per line)
    unsigned td1 = 0x7fffffffu;                          // tensor_dim1 (plenty of rows)
    unsigned st0 = (unsigned)stride_elems;               // tensor_dim0_stride
    i32x8 g1;
    g1[0] = (int)(1u << 16);                             // wg_mask=0 | data_size=1 (2B)
    g1[1] = (int)((td0 & 0xffffu) << 16);                // atomic_addr=0 | dim0[15:0]
    g1[2] = (int)((td0 >> 16) | ((td1 & 0xffffu) << 16));// dim0[31:16] | dim1[15:0]
    g1[3] = (int)((td1 >> 16) | ((unsigned)cols << 16)); // dim1[31:16] | tile_dim0
    g1[4] = (int)(unsigned)rows;                         // tile_dim1 | tile_dim2=0
    g1[5] = (int)st0;                                    // stride0[31:0]
    g1[6] = 0;                                           // stride0[47:32]=0 | stride1 lo=0
    g1[7] = 0;                                           // stride1 hi=0
    i32x4 gz4 = {0, 0, 0, 0};
    i32x8 gz8 = {0, 0, 0, 0, 0, 0, 0, 0};
    __builtin_amdgcn_tensor_load_to_lds(g0, g1, gz4, gz4, gz8, 0);
}

// ---------------------------------------------------------------- converts
__global__ void cvt_k(const float* __restrict__ s, bf16_t* __restrict__ d, int n) {
    int i = blockIdx.x * blockDim.x + threadIdx.x;
    if (i < n) d[i] = (bf16_t)s[i];
}

__global__ void cvt_pad_k(const float* __restrict__ s, bf16_t* __restrict__ d,
                          int rows_valid, int cols, int rows_total) {
    int i = blockIdx.x * blockDim.x + threadIdx.x;
    if (i < rows_total * cols) {
        int r = i / cols;
        d[i] = (r < rows_valid) ? (bf16_t)s[i] : (bf16_t)0.0f;
    }
}

// ---------------------------------------------------------------- RoPE (bf16 in-place)
__global__ void rope_k(bf16_t* __restrict__ T, const float* __restrict__ cs,
                       const float* __restrict__ sn) {
    int idx = blockIdx.x * blockDim.x + threadIdx.x;   // SEQ*NH*HALF
    if (idx >= SEQV * NH * HALFV) return;
    int i   = idx & (HALFV - 1);
    int hh  = (idx >> 6) & (NH - 1);
    int pos = idx >> 11;
    bf16_t* p = T + (size_t)pos * DIMV + hh * HD + 2 * i;
    float a = (float)p[0], b = (float)p[1];
    float c = cs[pos * HALFV + i], s = sn[pos * HALFV + i];
    p[0] = (bf16_t)(a * c - b * s);
    p[1] = (bf16_t)(a * s + b * c);
}

// ---------------------------------------------------------------- bf16 GEMM  C = A(MxK) * B(NxK)^T
// 128x128x32 double-buffered LDS tile. A staged via async-to-LDS (ASYNCcnt),
// B (weights) staged via TDM tensor_load_to_lds (TENSORcnt). 8 waves/block,
// each wave computes a 16x128 strip (8 WMMAs / k-step).
__global__ void gemm_bf16_k(const bf16_t* __restrict__ A, const bf16_t* __restrict__ B,
                            bf16_t* __restrict__ Cb, float* __restrict__ Cf,
                            int M, int N, int K) {
    __shared__ alignas(16) bf16_t As[2][128][32];
    __shared__ alignas(16) bf16_t Bs[2][128][32];

    const int t     = threadIdx.x;       // 0..255
    const int w     = t >> 5;
    const int lane  = t & 31;
    const int lhalf = lane >> 4;
    const int l15   = lane & 15;
    const int m0    = blockIdx.y * 128;
    const int n0    = blockIdx.x * 128;

    v8f acc[8];
#pragma unroll
    for (int j = 0; j < 8; ++j) acc[j] = v8f_zero();

    const int lr = t >> 1;            // 0..127 tile row this thread stages
    const int lc = (t & 1) * 16;      // 0 or 16
    const int gmA = m0 + lr;
    const bool aval = gmA < M;
    const bf16_t* aptr = A + (size_t)gmA * K + lc;
    const bf16_t* bbase = B + (size_t)n0 * K;

    // pre-zero A rows that are out of range (async loads are EXEC-masked off there)
    if (!aval) {
        u32x4 z = {};
        *(u32x4*)&As[0][lr][lc] = z;
        *(u32x4*)&As[0][lr][lc + 8] = z;
        *(u32x4*)&As[1][lr][lc] = z;
        *(u32x4*)&As[1][lr][lc + 8] = z;
    }

    // prologue: stage tile 0 into buffer 0
    if (w == 0)
        tdm_load_tile(bbase, (unsigned)(uintptr_t)&Bs[0][0][0], 128, 32, K);
    if (aval) {
        unsigned la = (unsigned)(uintptr_t)&As[0][lr][lc];
        async_b128(la, aptr);
        async_b128(la + 16, aptr + 8);
    }
    wait_async0();
    if (w == 0) __builtin_amdgcn_s_wait_tensorcnt(0);
    __syncthreads();

    int cur = 0;
    for (int k0 = 0; k0 < K; k0 += 32) {
        const int nxt = cur ^ 1;
        if (k0 + 32 < K) {
            // stage next tile while computing on current one
            if (w == 0)
                tdm_load_tile(bbase + k0 + 32, (unsigned)(uintptr_t)&Bs[nxt][0][0], 128, 32, K);
            if (aval) {
                unsigned la = (unsigned)(uintptr_t)&As[nxt][lr][lc];
                async_b128(la, aptr + k0 + 32);
                async_b128(la + 16, aptr + k0 + 40);
                if (k0 + 64 < K) __builtin_prefetch(aptr + k0 + 64, 0, 0);
            }
        }

        Frag af;
        {   // A operand: lane holds row m=l15, k = lhalf*16..+15 contiguous
            const bf16_t* p = &As[cur][w * 16 + l15][lhalf * 16];
            af.q[0] = *(const u32x4*)p;
            af.q[1] = *(const u32x4*)(p + 8);
        }
#pragma unroll
        for (int j = 0; j < 8; ++j) {
            Frag bf;  // B operand: lane holds col n=l15, k = lhalf*16..+15 contiguous
            const bf16_t* p = &Bs[cur][j * 16 + l15][lhalf * 16];
            bf.q[0] = *(const u32x4*)p;
            bf.q[1] = *(const u32x4*)(p + 8);
            acc[j] = wmma_bf16(af, bf, acc[j]);
        }

        wait_async0();
        if (w == 0) __builtin_amdgcn_s_wait_tensorcnt(0);
        __syncthreads();
        cur = nxt;
    }

#pragma unroll
    for (int j = 0; j < 8; ++j) {
        int gn = n0 + j * 16 + l15;
#pragma unroll
        for (int r = 0; r < 8; ++r) {
            int gm = m0 + w * 16 + r + lhalf * 8;
            if (gm < M) {
                if (Cf) Cf[(size_t)gm * N + gn] = acc[j][r];
                else    Cb[(size_t)gm * N + gn] = (bf16_t)acc[j][r];
            }
        }
    }
}

// ---------------------------------------------------------------- fused flash attention + prefix
// 1 wave = one 16-query tile of one head. block = 4 waves.
__global__ void attn_k(const bf16_t* __restrict__ Q, const bf16_t* __restrict__ Kc,
                       const bf16_t* __restrict__ V, const bf16_t* __restrict__ PK,
                       const bf16_t* __restrict__ PV, const float* __restrict__ pgate,
                       bf16_t* __restrict__ O) {
    __shared__ alignas(16) bf16_t VtS[4][HD * 32];   // per-wave V^T tile [d][key]
    __shared__ alignas(16) bf16_t PtS[4][16 * 32];   // per-wave P tile  [m][k]

    const int w     = threadIdx.x >> 5;
    const int lane  = threadIdx.x & 31;
    const int lhalf = lane >> 4;
    const int l15   = lane & 15;
    const int h     = blockIdx.y;
    const int q0    = (blockIdx.x * 4 + w) * 16;

    bf16_t* Vt = VtS[w];
    bf16_t* Pt = PtS[w];

    const float scale = 0.08838834764831845f;   // 1/sqrt(128)

    Frag qf[4];
    {
        const bf16_t* qbase = Q + (size_t)(q0 + l15) * DIMV + h * HD;
#pragma unroll
        for (int db = 0; db < 4; ++db) {
            const bf16_t* p = qbase + db * 32 + lhalf * 16;
            qf[db].q[0] = *(const u32x4*)p;
            qf[db].q[1] = *(const u32x4*)(p + 8);
        }
    }

    v8f acc[8];
#pragma unroll
    for (int j = 0; j < 8; ++j) acc[j] = v8f_zero();
    float mrow[8], lrow[8];
#pragma unroll
    for (int r = 0; r < 8; ++r) { mrow[r] = -1e30f; lrow[r] = 0.f; }

    // -------- causal flash loop over 32-key blocks --------
    for (int kb = 0; kb < q0 + 16; kb += 32) {
        {   // stage V^T into LDS: lane owns key kb+lane
            const bf16_t* vp = V + (size_t)(kb + lane) * DIMV + h * HD;
#pragma unroll
            for (int d8 = 0; d8 < 16; ++d8) {
                u32x4 dv = *(const u32x4*)(vp + d8 * 8);
                const bf16_t* e = (const bf16_t*)&dv;
#pragma unroll
                for (int ii = 0; ii < 8; ++ii)
                    Vt[(d8 * 8 + ii) * 32 + lane] = e[ii];
            }
        }
        asm volatile("" ::: "memory");

        v8f sc[2];
        sc[0] = v8f_zero(); sc[1] = v8f_zero();
#pragma unroll
        for (int nt = 0; nt < 2; ++nt) {
            const bf16_t* kbase = Kc + (size_t)(kb + nt * 16 + l15) * DIMV + h * HD;
#pragma unroll
            for (int db = 0; db < 4; ++db) {
                Frag bf;
                const bf16_t* p = kbase + db * 32 + lhalf * 16;
                bf.q[0] = *(const u32x4*)p;
                bf.q[1] = *(const u32x4*)(p + 8);
                sc[nt] = wmma_bf16(qf[db], bf, sc[nt]);
            }
        }

        const bool needmask = (kb + 31 > q0);
#pragma unroll
        for (int nt = 0; nt < 2; ++nt)
#pragma unroll
            for (int r = 0; r < 8; ++r) {
                float s = sc[nt][r] * scale;
                if (needmask) {
                    int n = kb + nt * 16 + l15;
                    int m = q0 + r + lhalf * 8;
                    if (n > m) s = -1e30f;
                }
                sc[nt][r] = s;
            }

#pragma unroll
        for (int r = 0; r < 8; ++r) {
            float v = hmax16(fmaxf(sc[0][r], sc[1][r]));
            float mnew = fmaxf(mrow[r], v);
            float corr = __expf(mrow[r] - mnew);
            mrow[r] = mnew;
            float p0 = __expf(sc[0][r] - mnew);
            float p1 = __expf(sc[1][r] - mnew);
            sc[0][r] = p0; sc[1][r] = p1;
            lrow[r] = lrow[r] * corr + hsum16(p0 + p1);
#pragma unroll
            for (int j = 0; j < 8; ++j) acc[j][r] *= corr;
        }

        // P -> LDS (C layout -> A layout relayout)
#pragma unroll
        for (int nt = 0; nt < 2; ++nt)
#pragma unroll
            for (int r = 0; r < 8; ++r)
                Pt[(r + lhalf * 8) * 32 + nt * 16 + l15] = (bf16_t)sc[nt][r];
        asm volatile("" ::: "memory");

        Frag pf;
        {
            const bf16_t* p = &Pt[l15 * 32 + lhalf * 16];
            pf.q[0] = *(const u32x4*)p;
            pf.q[1] = *(const u32x4*)(p + 8);
        }
#pragma unroll
        for (int j = 0; j < 8; ++j) {
            Frag bf;
            const bf16_t* p = &Vt[(j * 16 + l15) * 32 + lhalf * 16];
            bf.q[0] = *(const u32x4*)p;
            bf.q[1] = *(const u32x4*)(p + 8);
            acc[j] = wmma_bf16(pf, bf, acc[j]);
        }
        asm volatile("" ::: "memory");
    }

    // normalize causal output
#pragma unroll
    for (int r = 0; r < 8; ++r) {
        float inv = 1.0f / lrow[r];
#pragma unroll
        for (int j = 0; j < 8; ++j) acc[j][r] *= inv;
    }

    // -------- prefix attention (single 32-key block, keys >= 30 masked) --------
    {
        const bf16_t* vp = PV + (size_t)lane * DIMV + h * HD;
#pragma unroll
        for (int d8 = 0; d8 < 16; ++d8) {
            u32x4 dv = *(const u32x4*)(vp + d8 * 8);
            const bf16_t* e = (const bf16_t*)&dv;
#pragma unroll
            for (int ii = 0; ii < 8; ++ii)
                Vt[(d8 * 8 + ii) * 32 + lane] = e[ii];
        }
        asm volatile("" ::: "memory");

        v8f sc[2];
        sc[0] = v8f_zero(); sc[1] = v8f_zero();
#pragma unroll
        for (int nt = 0; nt < 2; ++nt) {
            const bf16_t* kbase = PK + (size_t)(nt * 16 + l15) * DIMV + h * HD;
#pragma unroll
            for (int db = 0; db < 4; ++db) {
                Frag bf;
                const bf16_t* p = kbase + db * 32 + lhalf * 16;
                bf.q[0] = *(const u32x4*)p;
                bf.q[1] = *(const u32x4*)(p + 8);
                sc[nt] = wmma_bf16(qf[db], bf, sc[nt]);
            }
        }

        float g = tanhf(pgate[h]);
#pragma unroll
        for (int nt = 0; nt < 2; ++nt)
#pragma unroll
            for (int r = 0; r < 8; ++r) {
                float s = sc[nt][r] * scale;
                int n = nt * 16 + l15;
                if (n >= PREV) s = -1e30f;
                sc[nt][r] = s;
            }

#pragma unroll
        for (int r = 0; r < 8; ++r) {
            float v = hmax16(fmaxf(sc[0][r], sc[1][r]));
            float p0 = __expf(sc[0][r] - v);
            float p1 = __expf(sc[1][r] - v);
            float rs = hsum16(p0 + p1);
            float ps = g / rs;                 // fold gate + 1/l into P
            sc[0][r] = p0 * ps;
            sc[1][r] = p1 * ps;
        }

#pragma unroll
        for (int nt = 0; nt < 2; ++nt)
#pragma unroll
            for (int r = 0; r < 8; ++r)
                Pt[(r + lhalf * 8) * 32 + nt * 16 + l15] = (bf16_t)sc[nt][r];
        asm volatile("" ::: "memory");

        Frag pf;
        {
            const bf16_t* p = &Pt[l15 * 32 + lhalf * 16];
            pf.q[0] = *(const u32x4*)p;
            pf.q[1] = *(const u32x4*)(p + 8);
        }
#pragma unroll
        for (int j = 0; j < 8; ++j) {
            Frag bf;
            const bf16_t* p = &Vt[(j * 16 + l15) * 32 + lhalf * 16];
            bf.q[0] = *(const u32x4*)p;
            bf.q[1] = *(const u32x4*)(p + 8);
            acc[j] = wmma_bf16(pf, bf, acc[j]);   // out += gate * pout
        }
    }

#pragma unroll
    for (int j = 0; j < 8; ++j) {
        int gn = h * HD + j * 16 + l15;
#pragma unroll
        for (int r = 0; r < 8; ++r) {
            int gm = q0 + r + lhalf * 8;
            O[(size_t)gm * DIMV + gn] = (bf16_t)acc[j][r];
        }
    }
}

// ---------------------------------------------------------------- host launch
extern "C" void kernel_launch(void* const* d_in, const int* in_sizes, int n_in,
                              void* d_out, int out_size, void* d_ws, size_t ws_size,
                              hipStream_t stream) {
    (void)in_sizes; (void)n_in; (void)out_size; (void)ws_size;

    const float* x      = (const float*)d_in[0];
    const float* prefix = (const float*)d_in[1];
    const float* pgate  = (const float*)d_in[2];
    const float* wq     = (const float*)d_in[3];
    const float* wk     = (const float*)d_in[4];
    const float* wv     = (const float*)d_in[5];
    const float* wo     = (const float*)d_in[6];
    const float* fcos   = (const float*)d_in[7];
    const float* fsin   = (const float*)d_in[8];
    float* out = (float*)d_out;

    char* ws = (char*)d_ws;
    size_t off = 0;
    auto salloc = [&](size_t bytes) -> void* {
        void* p = ws + off;
        off += (bytes + 255) & ~(size_t)255;
        return p;
    };
    bf16_t* xb    = (bf16_t*)salloc((size_t)SEQV * DIMV * 2);
    bf16_t* pfb   = (bf16_t*)salloc((size_t)PREP * DIMV * 2);
    bf16_t* wslot = (bf16_t*)salloc((size_t)DIMV * DIMV * 2);
    bf16_t* Qb    = (bf16_t*)salloc((size_t)SEQV * DIMV * 2);
    bf16_t* Kb    = (bf16_t*)salloc((size_t)SEQV * DIMV * 2);
    bf16_t* Vb    = (bf16_t*)salloc((size_t)SEQV * DIMV * 2);
    bf16_t* pkb   = (bf16_t*)salloc((size_t)PREP * DIMV * 2);
    bf16_t* pvb   = (bf16_t*)salloc((size_t)PREP * DIMV * 2);
    bf16_t* attnb = (bf16_t*)salloc((size_t)SEQV * DIMV * 2);

    const int NW = DIMV * DIMV;
    dim3 gfull(DIMV / 128, SEQV / 128);
    dim3 gpref(DIMV / 128, 1);

    cvt_k<<<(SEQV * DIMV + 255) / 256, 256, 0, stream>>>(x, xb, SEQV * DIMV);
    cvt_pad_k<<<(PREP * DIMV + 255) / 256, 256, 0, stream>>>(prefix, pfb, PREV, DIMV, PREP);

    // Q = x wq^T
    cvt_k<<<(NW + 255) / 256, 256, 0, stream>>>(wq, wslot, NW);
    gemm_bf16_k<<<gfull, 256, 0, stream>>>(xb, wslot, Qb, nullptr, SEQV, DIMV, DIMV);

    // K = x wk^T ; pk = prefix wk^T
    cvt_k<<<(NW + 255) / 256, 256, 0, stream>>>(wk, wslot, NW);
    gemm_bf16_k<<<gfull, 256, 0, stream>>>(xb, wslot, Kb, nullptr, SEQV, DIMV, DIMV);
    gemm_bf16_k<<<gpref, 256, 0, stream>>>(pfb, wslot, pkb, nullptr, PREP, DIMV, DIMV);

    // V = x wv^T ; pv = prefix wv^T
    cvt_k<<<(NW + 255) / 256, 256, 0, stream>>>(wv, wslot, NW);
    gemm_bf16_k<<<gfull, 256, 0, stream>>>(xb, wslot, Vb, nullptr, SEQV, DIMV, DIMV);
    gemm_bf16_k<<<gpref, 256, 0, stream>>>(pfb, wslot, pvb, nullptr, PREP, DIMV, DIMV);

    // RoPE on Q and K
    const int nrope = SEQV * NH * HALFV;
    rope_k<<<(nrope + 255) / 256, 256, 0, stream>>>(Qb, fcos, fsin);
    rope_k<<<(nrope + 255) / 256, 256, 0, stream>>>(Kb, fcos, fsin);

    // fused flash attention + gated prefix attention
    attn_k<<<dim3(SEQV / 16 / 4, NH), 128, 0, stream>>>(Qb, Kb, Vb, pkb, pvb, pgate, attnb);

    // out = attn wo^T  (f32 epilogue)
    cvt_k<<<(NW + 255) / 256, 256, 0, stream>>>(wo, wslot, NW);
    gemm_bf16_k<<<gfull, 256, 0, stream>>>(attnb, wslot, nullptr, out, SEQV, DIMV, DIMV);
}
